// GatedMultiHeadSelfAttention_33354716020744
// MI455X (gfx1250) — matrix operand order repaired and checked by reading, not compiled, not run on hardware
//
#include <hip/hip_runtime.h>

typedef __attribute__((ext_vector_type(16))) _Float16 v16h;
typedef __attribute__((ext_vector_type(8)))  _Float16 v8h;
typedef __attribute__((ext_vector_type(8)))  float    v8f;

#define B_  2
#define T_  2048
#define E_  1024
#define H_  16
#define D_  64
#define M_  (B_*T_)   /* 4096 */
#define N_  1024
#define K_  1024

__device__ inline v8f vzero8() {
  v8f z;
#pragma unroll
  for (int i = 0; i < 8; ++i) z[i] = 0.0f;
  return z;
}

// ---------------------------------------------------------------------------
// Flash attention: per (b,h), O = softmax(Q K^T / sqrt(D)) V
//   grid = (T/64, B*H), block = 128 (4 waves), 16 queries per wave.
//   Q,K: [B,H,T,D] f16   Vt: [B,H,D,T] f16   Out: [B*T, H*D] f16
// ---------------------------------------------------------------------------
__global__ __launch_bounds__(128, 2) void flash_attn(
    const _Float16* __restrict__ Q, const _Float16* __restrict__ K,
    const _Float16* __restrict__ Vt, _Float16* __restrict__ Out) {
  const int bh = blockIdx.y;            // 0..B*H-1
  const int b = bh >> 4, h = bh & (H_ - 1);
  const _Float16* Qp = Q  + (size_t)bh * T_ * D_;
  const _Float16* Kp = K  + (size_t)bh * T_ * D_;
  const _Float16* Vp = Vt + (size_t)bh * D_ * T_;

  const int wave = threadIdx.x >> 5, lane = threadIdx.x & 31;
  const int lr = lane & 15, hi = lane >> 4;
  const int q0 = blockIdx.x * 64 + wave * 16;

  __shared__ __attribute__((aligned(16))) _Float16 Pbuf[4][16 * 32];
  _Float16* pb = Pbuf[wave];

  // Q fragments (A-operand, 16 queries x K-dim 64 as two 16x32 frags) — resident
  v16h qf[2];
#pragma unroll
  for (int c = 0; c < 2; ++c) {
    const _Float16* p = Qp + (size_t)(q0 + lr) * D_ + c * 32 + hi * 8;
    union { v16h v; v8h h2[2]; } u;
    u.h2[0] = *(const v8h*)p;
    u.h2[1] = *(const v8h*)(p + 16);
    qf[c] = u.v;
  }

  float mrow[8], lrow[8];
  v8f o[4];
#pragma unroll
  for (int r = 0; r < 8; ++r) { mrow[r] = -1e30f; lrow[r] = 0.0f; }
#pragma unroll
  for (int tt = 0; tt < 4; ++tt) o[tt] = vzero8();

  const float sc = 0.125f * 1.44269504088896f;  // 1/sqrt(64) * log2(e); exp2 domain

#pragma unroll 1
  for (int kc = 0; kc < T_; kc += 32) {
    // ---- S = Q K^T : two 16x16 tiles (keys kc..kc+15, kc+16..kc+31) ----
    v8f s[2];
    s[0] = vzero8();
    s[1] = vzero8();
#pragma unroll
    for (int j = 0; j < 2; ++j) {
      const int n0 = kc + j * 16;
#pragma unroll
      for (int c = 0; c < 2; ++c) {
        // K fragment (B-operand): rows = keys, contiguous K-dim = d
        const _Float16* p = Kp + (size_t)(n0 + lr) * D_ + c * 32 + hi * 8;
        union { v16h v; v8h h2[2]; } u;
        u.h2[0] = *(const v8h*)p;
        u.h2[1] = *(const v8h*)(p + 16);
        s[j] = __builtin_amdgcn_wmma_f32_16x16x32_f16(
            false, qf[c], false, u.v, (short)0, s[j], false, false);
      }
    }

    // ---- online softmax over this 32-key chunk (row r lives in reg r) ----
#pragma unroll
    for (int r = 0; r < 8; ++r) {
      float v0 = s[0][r] * sc;
      float v1 = s[1][r] * sc;
      float a = fmaxf(v0, v1);
      a = fmaxf(a, __shfl_xor(a, 1, 32));
      a = fmaxf(a, __shfl_xor(a, 2, 32));
      a = fmaxf(a, __shfl_xor(a, 4, 32));
      a = fmaxf(a, __shfl_xor(a, 8, 32));
      const float mnew = fmaxf(mrow[r], a);
      const float p0 = exp2f(v0 - mnew);
      const float p1 = exp2f(v1 - mnew);
      float rs = p0 + p1;
      rs += __shfl_xor(rs, 1, 32);
      rs += __shfl_xor(rs, 2, 32);
      rs += __shfl_xor(rs, 4, 32);
      rs += __shfl_xor(rs, 8, 32);
      const float corr = exp2f(mrow[r] - mnew);
      lrow[r] = lrow[r] * corr + rs;
      mrow[r] = mnew;
#pragma unroll
      for (int tt = 0; tt < 4; ++tt) o[tt][r] *= corr;
      // stash probabilities (row q = r + 8*hi, keys j*16 + lr)
      pb[(r + hi * 8) * 32 + lr]      = (_Float16)p0;
      pb[(r + hi * 8) * 32 + 16 + lr] = (_Float16)p1;
    }

    // ---- P fragment (A-operand 16x32) from per-wave LDS ----
    union { v16h v; v8h h2[2]; } up;
    up.h2[0] = *(const v8h*)&pb[lr * 32 + hi * 8];
    up.h2[1] = *(const v8h*)&pb[lr * 32 + 16 + hi * 8];

    // ---- O += P * V : four 16x16 d-tiles, B-operand from Vt (K-major keys) ----
#pragma unroll
    for (int tt = 0; tt < 4; ++tt) {
      const _Float16* p = Vp + (size_t)(tt * 16 + lr) * T_ + kc + hi * 8;
      union { v16h v; v8h h2[2]; } u;
      u.h2[0] = *(const v8h*)p;
      u.h2[1] = *(const v8h*)(p + 16);
      o[tt] = __builtin_amdgcn_wmma_f32_16x16x32_f16(
          false, up.v, false, u.v, (short)0, o[tt], false, false);
    }
  }

  // ---- normalize and store: Out[(b*T + q), h*D + d] ----
#pragma unroll
  for (int r = 0; r < 8; ++r) {
    const float inv = 1.0f / lrow[r];
    const int q = q0 + r + hi * 8;
#pragma unroll
    for (int tt = 0; tt < 4; ++tt) {
      const int d = tt * 16 + lr;
      Out[((size_t)(b * T_ + q)) * (H_ * D_) + h * D_ + d] = (_Float16)(o[tt][r] * inv);
    }
  }
}

// ---------------------------------------------------------------------------
// WMMA GEMM: C[M_ x N_] = A[M_ x K_] * Bt[N_ x K_]^T + bias
//   128x128 block tile, 256 threads (8 waves), k-step 32.
//   Tiles staged with GLOBAL_LOAD_ASYNC_TO_LDS_B128 (ASYNCcnt path, no VGPR data).
//   MODE 0: f16 row-major   MODE 1: f16 scatter to [B,H,T,D]
//   MODE 2: f32 row-major   MODE 3: f16 scatter to [B,H,D,T] (V transposed)
// ---------------------------------------------------------------------------
template <int MODE>
__global__ __launch_bounds__(256, 2) void wmma_gemm(
    const _Float16* __restrict__ A, const _Float16* __restrict__ Bt,
    const float* __restrict__ bias, void* __restrict__ Cout) {
  __shared__ __attribute__((aligned(16))) _Float16 As[128 * 32];
  __shared__ __attribute__((aligned(16))) _Float16 Bs[128 * 32];

  const int m0 = blockIdx.y * 128;
  const int n0 = blockIdx.x * 128;
  const int t = threadIdx.x;
  const int lane = t & 31;
  const int wave = t >> 5;
  const int lr = lane & 15, hi = lane >> 4;
  const int wm = wave >> 2;        // 0..1  -> 64 output rows per wave
  const int wn = wave & 3;         // 0..3  -> 32 output cols per wave
  const int srow = t >> 1;         // 0..127 staging row
  const int spos = (t & 1) << 4;   // 0 or 16 staging half-offset

  // LDS byte addresses for this thread's staging slots (aperture keeps the
  // LDS offset in the low 32 bits of the generic pointer).
  const unsigned la0 = (unsigned)(size_t)&As[srow * 32 + spos];
  const unsigned lb0 = (unsigned)(size_t)&Bs[srow * 32 + spos];

  v8f acc[4][2];
#pragma unroll
  for (int i = 0; i < 4; ++i)
#pragma unroll
    for (int j = 0; j < 2; ++j) acc[i][j] = vzero8();

#pragma unroll 1
  for (int k0 = 0; k0 < K_; k0 += 32) {
    // ---- async global -> LDS staging (32 bytes per thread per matrix) ----
    {
      unsigned long long ga =
          (unsigned long long)(size_t)(A  + (size_t)(m0 + srow) * K_ + k0 + spos);
      unsigned long long gb =
          (unsigned long long)(size_t)(Bt + (size_t)(n0 + srow) * K_ + k0 + spos);
      asm volatile("global_load_async_to_lds_b128 %0, %1, off"
                   :: "v"(la0), "v"(ga) : "memory");
      asm volatile("global_load_async_to_lds_b128 %0, %1, off offset:16"
                   :: "v"(la0), "v"(ga) : "memory");
      asm volatile("global_load_async_to_lds_b128 %0, %1, off"
                   :: "v"(lb0), "v"(gb) : "memory");
      asm volatile("global_load_async_to_lds_b128 %0, %1, off offset:16"
                   :: "v"(lb0), "v"(gb) : "memory");
      asm volatile("s_wait_asynccnt 0x0" ::: "memory");
    }
    __syncthreads();

    v16h af[4], bf[2];
#pragma unroll
    for (int i = 0; i < 4; ++i) {
      int r0 = wm * 64 + i * 16;
      union { v16h v; v8h h[2]; } u;
      u.h[0] = *(const v8h*)&As[(r0 + lr) * 32 + hi * 8];
      u.h[1] = *(const v8h*)&As[(r0 + lr) * 32 + 16 + hi * 8];
      af[i] = u.v;
    }
#pragma unroll
    for (int j = 0; j < 2; ++j) {
      int c0 = wn * 32 + j * 16;
      union { v16h v; v8h h[2]; } u;
      u.h[0] = *(const v8h*)&Bs[(c0 + lr) * 32 + hi * 8];
      u.h[1] = *(const v8h*)&Bs[(c0 + lr) * 32 + 16 + hi * 8];
      bf[j] = u.v;
    }
#pragma unroll
    for (int i = 0; i < 4; ++i)
#pragma unroll
      for (int j = 0; j < 2; ++j)
        acc[i][j] = __builtin_amdgcn_wmma_f32_16x16x32_f16(
            false, af[i], false, bf[j], (short)0, acc[i][j], false, false);
    __syncthreads();
  }

  // Epilogue: C/D layout -> element (row = mt + r + 8*hi, col = nt + lr) in reg r
#pragma unroll
  for (int i = 0; i < 4; ++i) {
#pragma unroll
    for (int j = 0; j < 2; ++j) {
      const int mt = m0 + wm * 64 + i * 16;
      const int nt = n0 + wn * 32 + j * 16;
      const int col = nt + lr;
      const float bv = bias[col];
#pragma unroll
      for (int r = 0; r < 8; ++r) {
        const int row = mt + r + hi * 8;
        const float v = acc[i][j][r] + bv;
        if constexpr (MODE == 2) {
          ((float*)Cout)[(size_t)row * N_ + col] = v;
        } else if constexpr (MODE == 0) {
          ((_Float16*)Cout)[(size_t)row * N_ + col] = (_Float16)v;
        } else {
          const int b = row >> 11, tt = row & (T_ - 1);   // row = b*T + t
          const int h = col >> 6,  d  = col & (D_ - 1);   // col = h*D + d
          if constexpr (MODE == 1)   // [B,H,T,D]
            ((_Float16*)Cout)[(((size_t)(b * H_ + h)) * T_ + tt) * D_ + d] = (_Float16)v;
          else                       // MODE 3: [B,H,D,T]
            ((_Float16*)Cout)[(((size_t)(b * H_ + h)) * D_ + d) * T_ + tt] = (_Float16)v;
        }
      }
    }
  }
}

// ---------------------------------------------------------------------------
// Pack / convert kernels
// ---------------------------------------------------------------------------
__global__ void convert_f32_to_f16(const float* __restrict__ in,
                                   _Float16* __restrict__ out) {
  int i = blockIdx.x * 256 + threadIdx.x;
  out[i] = (_Float16)in[i];
}

// W: [H, E, D] fp32  ->  Wt: [n = h*D+d][e] f16   (K-major rows, K = E)
__global__ void pack_w_ed(const float* __restrict__ W, _Float16* __restrict__ Wt) {
  int idx = blockIdx.x * 256 + threadIdx.x;     // over N_*K_ = 1M
  int n = idx >> 10, e = idx & 1023;
  int h = n >> 6, d = n & 63;
  Wt[idx] = (_Float16)W[((size_t)h * E_ + e) * D_ + d];
}

// Wo: [H, D, E] fp32 -> Wt: [e][n = h*D+d] f16 with gate folded (K-major, K = H*D)
__global__ void pack_wo(const float* __restrict__ Wo, const float* __restrict__ gate,
                        _Float16* __restrict__ Wt) {
  int idx = blockIdx.x * 256 + threadIdx.x;     // over N_*K_ = 1M
  int e = idx >> 10, n = idx & 1023;
  int h = n >> 6, d = n & 63;
  Wt[idx] = (_Float16)(gate[h] * Wo[((size_t)h * D_ + d) * E_ + e]);
}

// bo_g[e] = sum_h gate[h] * bo[h][e]
__global__ void make_bo(const float* __restrict__ bo, const float* __restrict__ gate,
                        float* __restrict__ out) {
  int e = blockIdx.x * 256 + threadIdx.x;       // 1024
  float s = 0.0f;
  for (int h = 0; h < H_; ++h) s += gate[h] * bo[(size_t)h * E_ + e];
  out[e] = s;
}

// ---------------------------------------------------------------------------
// Host launcher
// ---------------------------------------------------------------------------
extern "C" void kernel_launch(void* const* d_in, const int* in_sizes, int n_in,
                              void* d_out, int out_size, void* d_ws, size_t ws_size,
                              hipStream_t stream) {
  (void)in_sizes; (void)n_in; (void)out_size; (void)ws_size;
  const float* X    = (const float*)d_in[0];
  const float* Wq   = (const float*)d_in[1];
  const float* bq   = (const float*)d_in[2];
  const float* Wk   = (const float*)d_in[3];
  const float* bk   = (const float*)d_in[4];
  const float* Wv   = (const float*)d_in[5];
  const float* bv   = (const float*)d_in[6];
  const float* Wo   = (const float*)d_in[7];
  const float* bo   = (const float*)d_in[8];
  const float* gate = (const float*)d_in[9];

  char* p = (char*)d_ws;
  auto take = [&](size_t bytes) -> char* {
    char* r = p;
    p += (bytes + 255) & ~(size_t)255;
    return r;
  };
  _Float16* Xh  = (_Float16*)take((size_t)M_ * K_ * 2);   // hidden f16 [M,K]
  _Float16* Wqt = (_Float16*)take((size_t)N_ * K_ * 2);
  _Float16* Wkt = (_Float16*)take((size_t)N_ * K_ * 2);
  _Float16* Wvt = (_Float16*)take((size_t)N_ * K_ * 2);
  _Float16* Wot = (_Float16*)take((size_t)N_ * K_ * 2);
  float*    bog = (float*)take((size_t)E_ * 4);
  _Float16* Qb  = (_Float16*)take((size_t)M_ * N_ * 2);   // [B,H,T,D]
  _Float16* Kb  = (_Float16*)take((size_t)M_ * N_ * 2);   // [B,H,T,D]
  _Float16* Vtb = (_Float16*)take((size_t)M_ * N_ * 2);   // [B,H,D,T]
  _Float16* A2  = (_Float16*)take((size_t)M_ * N_ * 2);   // attn out [M, H*D]

  convert_f32_to_f16<<<(M_ * K_) / 256, 256, 0, stream>>>(X, Xh);
  pack_w_ed<<<(N_ * K_) / 256, 256, 0, stream>>>(Wq, Wqt);
  pack_w_ed<<<(N_ * K_) / 256, 256, 0, stream>>>(Wk, Wkt);
  pack_w_ed<<<(N_ * K_) / 256, 256, 0, stream>>>(Wv, Wvt);
  pack_wo<<<(N_ * K_) / 256, 256, 0, stream>>>(Wo, gate, Wot);
  make_bo<<<E_ / 256, 256, 0, stream>>>(bo, gate, bog);

  dim3 gg(N_ / 128, M_ / 128);   // (8, 32)
  wmma_gemm<1><<<gg, 256, 0, stream>>>(Xh, Wqt, bq, Qb);
  wmma_gemm<1><<<gg, 256, 0, stream>>>(Xh, Wkt, bk, Kb);
  wmma_gemm<3><<<gg, 256, 0, stream>>>(Xh, Wvt, bv, Vtb);

  flash_attn<<<dim3(T_ / 64, B_ * H_), 128, 0, stream>>>(Qb, Kb, Vtb, A2);

  wmma_gemm<2><<<gg, 256, 0, stream>>>(A2, Wot, bog, d_out);
}
